// TFRelPartialLearnableMultiHeadAttn_41377714930138
// MI455X (gfx1250) — matrix operand order repaired and checked by reading, compiled, not run on hardware
//
#include <hip/hip_runtime.h>

// ---------------------------------------------------------------------------
// Transformer-XL rel-attention block for gfx1250 (MI455X), bf16 WMMA pipeline.
// All GEMM B-operands stored N-major ("Bt") so WMMA B-fragments are two
// contiguous b128 LDS loads per lane; staging uses async global->LDS copies.
// ---------------------------------------------------------------------------

#define QLEN 1024
#define MLEN 1024
#define KLENC 2048
#define BSZ  4
#define NH   16
#define DH   64
#define DM   1024
#define ATT_SCALE 0.125f

#define ASYNC_LDS 1

typedef __attribute__((ext_vector_type(16))) __bf16 v16bf;
typedef __attribute__((ext_vector_type(8)))  float  v8f;

__device__ __forceinline__ unsigned short f2bf(float f) {
  unsigned int u = __float_as_uint(f);
  u += 0x7fffu + ((u >> 16) & 1u);           // round-to-nearest-even
  return (unsigned short)(u >> 16);
}
__device__ __forceinline__ float bf2f(unsigned short h) {
  return __uint_as_float(((unsigned int)h) << 16);
}

__device__ __forceinline__ v8f wmma_bf16(v16bf a, v16bf b, v8f c) {
  return __builtin_amdgcn_wmma_f32_16x16x32_bf16(false, a, false, b, (short)0, c,
                                                 false, false);
}

// --- async global->LDS staging (ASYNCcnt path), with synchronous fallback ---
__device__ __forceinline__ void stage16(unsigned short* lds_dst,
                                        const unsigned short* gsrc) {
#if ASYNC_LDS
  const unsigned ldsoff = (unsigned)(unsigned long long)lds_dst;  // aperture low 32b
  asm volatile("global_load_async_to_lds_b128 %0, %1, off"
               :: "v"(ldsoff), "v"(gsrc)
               : "memory");
#else
  *(uint4*)lds_dst = *(const uint4*)gsrc;
#endif
}
__device__ __forceinline__ void stage_wait() {
#if ASYNC_LDS
#if defined(__has_builtin)
#if __has_builtin(__builtin_amdgcn_s_wait_asynccnt)
  __builtin_amdgcn_s_wait_asynccnt(0);
#else
  asm volatile("s_wait_asynccnt 0x0" ::: "memory");
#endif
#else
  asm volatile("s_wait_asynccnt 0x0" ::: "memory");
#endif
#endif
}

// A-fragment (16 rows x 32 K, bf16) from row-major ushort buffer (LDS/global).
// lane l -> m = l%16, kh = l/16; vgpr v -> K = 16*(v/4)+8*kh+2*(v&3):
// two contiguous 8-ushort runs -> 2x b128.
__device__ __forceinline__ v16bf load_a_frag(const unsigned short* p, int ld,
                                             int m0, int kb, int lane) {
  union { v16bf v; unsigned int u[8]; } f;
  const int m  = m0 + (lane & 15);
  const int kh = (lane >> 4) & 1;
#pragma unroll
  for (int vv = 0; vv < 8; ++vv) {
    const int k = kb + ((vv >> 2) << 4) + (kh << 3) + ((vv & 3) << 1);
    f.u[vv] = *(const unsigned int*)(p + (size_t)m * ld + k);
  }
  return f.v;
}

// B-fragment from an N-major ("B transposed", N x K row-major) buffer.
// lane l -> col n = n0 + l%16, kh = l/16; vgpr v -> K = 16*(v/4)+8*kh+2*(v&3):
// two contiguous 8-ushort runs -> 2x b128.
__device__ __forceinline__ v16bf load_bt_frag(const unsigned short* p, int ld,
                                              int kb, int n0, int lane) {
  union { v16bf v; unsigned int u[8]; } f;
  const int n  = n0 + (lane & 15);
  const int kh = (lane >> 4) & 1;
  const unsigned short* row = p + (size_t)n * ld + kb + (kh << 3);
#pragma unroll
  for (int vv = 0; vv < 8; ++vv) f.u[vv] = *(const unsigned int*)(row + ((vv >> 2) << 4) + ((vv & 3) << 1));
  return f.v;
}

// ---------------------------------------------------------------------------
// 1a) f32 -> bf16 convert (vectorized x4).
// ---------------------------------------------------------------------------
__global__ __launch_bounds__(256) void k_f32_to_bf16(const float* __restrict__ src,
                                                     unsigned short* __restrict__ dst,
                                                     int n4) {
  const int i = blockIdx.x * 256 + threadIdx.x;
  if (i >= n4) return;
  const float4 v = ((const float4*)src)[i];
  ((ushort4*)dst)[i] = make_ushort4(f2bf(v.x), f2bf(v.y), f2bf(v.z), f2bf(v.w));
}

// 1b) f32 (K x N) -> bf16 transposed (N x K): weight matrices become Bt layout.
__global__ __launch_bounds__(256) void k_f32_to_bf16_t(const float* __restrict__ src,
                                                       unsigned short* __restrict__ dst,
                                                       int K, int N) {
  const int t = blockIdx.x * 256 + threadIdx.x;
  const int n = t % N, k = t / N;                 // coalesced reads
  if (k >= K) return;
  dst[(size_t)n * K + k] = f2bf(src[t]);
}

// ---------------------------------------------------------------------------
// 2) Tiled bf16 GEMM: C[MxN] = A[MxK] * B[KxN], A row-major, B given N-major.
//    128x128 block, 256 threads (8 waves, 2x4 wave grid, 64x32 per wave).
// ---------------------------------------------------------------------------
template <int N, int K, bool F32OUT>
__global__ __launch_bounds__(256) void k_gemm_bf16(const unsigned short* __restrict__ A,
                                                   const unsigned short* __restrict__ Bt,
                                                   void* __restrict__ C) {
  __shared__ __align__(16) unsigned short As[128 * 32];   // (m, k)
  __shared__ __align__(16) unsigned short Bts[128 * 32];  // (n, k)
  const int tid = threadIdx.x, lane = tid & 31, wv = tid >> 5;
  const int wm = (wv >> 2) * 64;   // 0 or 64
  const int wn = (wv & 3) * 32;    // 0,32,64,96
  const int m0b = blockIdx.y * 128, n0b = blockIdx.x * 128;

  v8f acc[4][2];
#pragma unroll
  for (int a = 0; a < 4; ++a)
#pragma unroll
    for (int b = 0; b < 2; ++b)
#pragma unroll
      for (int r = 0; r < 8; ++r) acc[a][b][r] = 0.0f;

  for (int k0 = 0; k0 < K; k0 += 32) {
#pragma unroll
    for (int c = 0; c < 2; ++c) {
      const int idx = tid + c * 256;                     // 512 chunks each
      const int ra = idx >> 2, pa = (idx & 3) * 8;       // 128 rows x 4 chunks
      stage16(As + ra * 32 + pa, A + (size_t)(m0b + ra) * K + k0 + pa);
      stage16(Bts + ra * 32 + pa, Bt + (size_t)(n0b + ra) * K + k0 + pa);
    }
    stage_wait();
    __syncthreads();
    v16bf af[4], bfr[2];
#pragma unroll
    for (int mt = 0; mt < 4; ++mt) af[mt] = load_a_frag(As, 32, wm + mt * 16, 0, lane);
#pragma unroll
    for (int nt = 0; nt < 2; ++nt) bfr[nt] = load_bt_frag(Bts, 32, 0, wn + nt * 16, lane);
#pragma unroll
    for (int mt = 0; mt < 4; ++mt)
#pragma unroll
      for (int nt = 0; nt < 2; ++nt) acc[mt][nt] = wmma_bf16(af[mt], bfr[nt], acc[mt][nt]);
    __syncthreads();
  }

  const int rbase = (lane >> 4) << 3;
  const int cn = lane & 15;
#pragma unroll
  for (int mt = 0; mt < 4; ++mt)
#pragma unroll
    for (int nt = 0; nt < 2; ++nt) {
      const int col = n0b + wn + nt * 16 + cn;
#pragma unroll
      for (int r = 0; r < 8; ++r) {
        const int row = m0b + wm + mt * 16 + rbase + r;
        if (F32OUT) ((float*)C)[(size_t)row * N + col] = acc[mt][nt][r];
        else ((unsigned short*)C)[(size_t)row * N + col] = f2bf(acc[mt][nt][r]);
      }
    }
}

// ---------------------------------------------------------------------------
// 3) Repack w_heads -> Q(+biases) (i,d), K (j,d), V^T (d,j) per (b, head).
// ---------------------------------------------------------------------------
__global__ __launch_bounds__(256) void k_repack(const unsigned short* __restrict__ wheads,
                                                const float* __restrict__ rwb,
                                                const float* __restrict__ rrb,
                                                unsigned short* __restrict__ Qrw,
                                                unsigned short* __restrict__ Qrr,
                                                unsigned short* __restrict__ Kn,
                                                unsigned short* __restrict__ Vt) {
  const int t  = blockIdx.x * 256 + threadIdx.x;   // KLENC*BSZ*NH*DH = 8388608
  const int hd = t & 1023;
  const int b  = (t >> 10) & 3;
  const int j  = t >> 12;
  const int n  = hd >> 6, d = hd & 63;
  const int bn = b * NH + n;
  const size_t src = (size_t)(j * BSZ + b) * 3072;

  Kn[((size_t)bn * KLENC + j) * DH + d] = wheads[src + 1024 + hd];
  Vt[((size_t)bn * DH + d) * KLENC + j] = wheads[src + 2048 + hd];
  if (j >= MLEN) {
    const int i = j - MLEN;
    const float q = bf2f(wheads[src + hd]);
    const size_t qi = ((size_t)bn * QLEN + i) * DH + d;
    Qrw[qi] = f2bf(q + rwb[hd]);
    Qrr[qi] = f2bf(q + rrb[hd]);
  }
}

// ---------------------------------------------------------------------------
// 4) Flash attention with exact rel-shift:
//    BD[i,j] = qrr[i] . rhk[j - i + QLEN-1]  (valid wherever mask allows).
//    Block: 64 i-rows of one (b,n); 4 waves x 16-row tiles; j streamed by 32.
// ---------------------------------------------------------------------------
__global__ __launch_bounds__(128) void k_flash_attn(const unsigned short* __restrict__ Qrw,
                                                    const unsigned short* __restrict__ Qrr,
                                                    const unsigned short* __restrict__ Kn,
                                                    const unsigned short* __restrict__ Vt,
                                                    const unsigned short* __restrict__ rhk,
                                                    unsigned short* __restrict__ avec) {
  __shared__ __align__(16) unsigned short Knl[32 * 64];   // (j-local, d)
  __shared__ __align__(16) unsigned short Vtl[64 * 32];   // (d, j-local)
  __shared__ __align__(16) unsigned short Rl[96 * 64];    // (rel band row, d)
  __shared__ __align__(16) unsigned short Pl[4 * 16 * 32];

  const int tid = threadIdx.x, lane = tid & 31, wv = tid >> 5;
  const int i0b = blockIdx.x * 64;
  const int n = blockIdx.y, b = blockIdx.z;
  const int bn = b * NH + n;
  const int woff = wv * 16;
  const int i0w = i0b + woff;

  const unsigned short* KnG  = Kn + (size_t)bn * KLENC * DH;
  const unsigned short* VtG  = Vt + (size_t)bn * DH * KLENC;
  const unsigned short* RG   = rhk + n * DH;              // row rel, cols n*64..
  const unsigned short* QrwG = Qrw + ((size_t)bn * QLEN + i0w) * DH;
  const unsigned short* QrrG = Qrr + ((size_t)bn * QLEN + i0w) * DH;
  unsigned short* Plw = Pl + wv * 512;

  const v16bf aw0 = load_a_frag(QrwG, DH, 0, 0, lane);
  const v16bf aw1 = load_a_frag(QrwG, DH, 0, 32, lane);
  const v16bf ar0 = load_a_frag(QrrG, DH, 0, 0, lane);
  const v16bf ar1 = load_a_frag(QrrG, DH, 0, 32, lane);

  v8f o[4];
  float mstat[8], lstat[8];
#pragma unroll
  for (int dt = 0; dt < 4; ++dt)
#pragma unroll
    for (int r = 0; r < 8; ++r) o[dt][r] = 0.0f;
#pragma unroll
  for (int r = 0; r < 8; ++r) { mstat[r] = -1e30f; lstat[r] = 0.0f; }

  const int xn0 = 48 - woff;          // wave's 48-wide window inside the 96 band
  const int rbase = (lane >> 4) << 3;
  const int jj15 = lane & 15;
  const int T = (i0b + 64 + MLEN + 31) >> 5;

  for (int jt = 0; jt < T; ++jt) {
    const int j0 = jt * 32;
    const int relbase = j0 + 960 - i0b;   // band row cc -> rel = relbase + cc

    // --- stage K slice (j,d), V^T slice (d,j), rel band (rel,d) into LDS ---
#pragma unroll
    for (int c = 0; c < 2; ++c) {
      const int idx = tid + c * 128;
      const int rK = idx >> 3, pK = (idx & 7) * 8;   // Knl: 32 rows x 8 chunks
      stage16(Knl + rK * 64 + pK, KnG + (size_t)(j0 + rK) * DH + pK);
      const int dV = idx >> 2, pV = (idx & 3) * 8;   // Vtl: 64 rows x 4 chunks
      stage16(Vtl + dV * 32 + pV, VtG + (size_t)dV * KLENC + j0 + pV);
    }
#pragma unroll
    for (int c = 0; c < 6; ++c) {
      const int idx = tid + c * 128;
      const int cc = idx >> 3, pR = (idx & 7) * 8;   // Rl: 96 rows x 8 chunks
      const int rel = relbase + cc;
      if (rel < KLENC) stage16(Rl + cc * 64 + pR, RG + (size_t)rel * DM + pR);
      else *(uint4*)(Rl + cc * 64 + pR) = make_uint4(0u, 0u, 0u, 0u);
    }
    if (jt + 1 < T) {
      __builtin_prefetch(KnG + (size_t)(j0 + 32 + (tid & 31)) * DH, 0, 1);
      __builtin_prefetch(VtG + (size_t)(tid & 63) * KLENC + j0 + 32, 0, 1);
    }
    stage_wait();
    __syncthreads();

    // --- AC (2x16 j-cols) and X band (3x16 rel-cols), K=64 -> 2 k-steps ---
    v8f sv0, sv1, xv0, xv1, xv2;
#pragma unroll
    for (int r = 0; r < 8; ++r) { sv0[r] = 0; sv1[r] = 0; xv0[r] = 0; xv1[r] = 0; xv2[r] = 0; }

    sv0 = wmma_bf16(aw0, load_bt_frag(Knl, 64, 0, 0, lane), sv0);
    sv0 = wmma_bf16(aw1, load_bt_frag(Knl, 64, 32, 0, lane), sv0);
    sv1 = wmma_bf16(aw0, load_bt_frag(Knl, 64, 0, 16, lane), sv1);
    sv1 = wmma_bf16(aw1, load_bt_frag(Knl, 64, 32, 16, lane), sv1);

    xv0 = wmma_bf16(ar0, load_bt_frag(Rl, 64, 0, xn0, lane), xv0);
    xv0 = wmma_bf16(ar1, load_bt_frag(Rl, 64, 32, xn0, lane), xv0);
    xv1 = wmma_bf16(ar0, load_bt_frag(Rl, 64, 0, xn0 + 16, lane), xv1);
    xv1 = wmma_bf16(ar1, load_bt_frag(Rl, 64, 32, xn0 + 16, lane), xv1);
    xv2 = wmma_bf16(ar0, load_bt_frag(Rl, 64, 0, xn0 + 32, lane), xv2);
    xv2 = wmma_bf16(ar1, load_bt_frag(Rl, 64, 32, xn0 + 32, lane), xv2);

    // --- rel-shift gather (width-16 shuffles), mask, online softmax ---
    float corr8[8];
#pragma unroll
    for (int r = 0; r < 8; ++r) {
      const int ii = rbase + r;
      float s0, s1;
      {
        const int c = jj15 - ii + 15;                   // [0,30]
        const int idx = c & 15;
        const float g0 = __shfl(xv0[r], idx, 16);
        const float g1 = __shfl(xv1[r], idx, 16);
        const float bd = (c < 16) ? g0 : g1;
        const float val = (sv0[r] + bd) * ATT_SCALE;
        s0 = ((j0 + jj15) > (i0w + ii + MLEN)) ? -1e30f : val;
      }
      {
        const int c = jj15 - ii + 31;                   // [16,46]
        const int idx = c & 15;
        const float g1 = __shfl(xv1[r], idx, 16);
        const float g2 = __shfl(xv2[r], idx, 16);
        const float bd = (c < 32) ? g1 : g2;
        const float val = (sv1[r] + bd) * ATT_SCALE;
        s1 = ((j0 + 16 + jj15) > (i0w + ii + MLEN)) ? -1e30f : val;
      }
      float rm = fmaxf(s0, s1);
      rm = fmaxf(rm, __shfl_xor(rm, 1, 16));
      rm = fmaxf(rm, __shfl_xor(rm, 2, 16));
      rm = fmaxf(rm, __shfl_xor(rm, 4, 16));
      rm = fmaxf(rm, __shfl_xor(rm, 8, 16));
      const float mnew = fmaxf(mstat[r], rm);
      const float corr = __expf(mstat[r] - mnew);
      mstat[r] = mnew;
      const float p0 = __expf(s0 - mnew);
      const float p1 = __expf(s1 - mnew);
      float ls = p0 + p1;
      ls += __shfl_xor(ls, 1, 16);
      ls += __shfl_xor(ls, 2, 16);
      ls += __shfl_xor(ls, 4, 16);
      ls += __shfl_xor(ls, 8, 16);
      lstat[r] = lstat[r] * corr + ls;
      corr8[r] = corr;
      Plw[ii * 32 + jj15] = f2bf(p0);
      Plw[ii * 32 + 16 + jj15] = f2bf(p1);
    }

    // --- rescale O, then P (16x32) @ V (32x64) via WMMA on V^T ---
#pragma unroll
    for (int dt = 0; dt < 4; ++dt)
#pragma unroll
      for (int r = 0; r < 8; ++r) o[dt][r] *= corr8[r];
    const v16bf pa = load_a_frag(Plw, 32, 0, 0, lane);
#pragma unroll
    for (int dt = 0; dt < 4; ++dt)
      o[dt] = wmma_bf16(pa, load_bt_frag(Vtl, 32, 0, dt * 16, lane), o[dt]);
    __syncthreads();
  }

  // epilogue: normalize and emit attn_vec rows (i*BSZ + b, NH*DH)
#pragma unroll
  for (int r = 0; r < 8; ++r) lstat[r] = 1.0f / lstat[r];
#pragma unroll
  for (int dt = 0; dt < 4; ++dt)
#pragma unroll
    for (int r = 0; r < 8; ++r) {
      const int i = i0w + rbase + r;
      avec[((size_t)i * BSZ + b) * DM + n * DH + dt * 16 + jj15] =
          f2bf(o[dt][r] * lstat[r]);
    }
}

// ---------------------------------------------------------------------------
// 5) Residual + LayerNorm, one block per row of 1024.
// ---------------------------------------------------------------------------
__global__ __launch_bounds__(256) void k_ln(const float* __restrict__ wres,
                                            const float* __restrict__ ao,
                                            const float* __restrict__ gamma,
                                            const float* __restrict__ beta,
                                            float* __restrict__ out) {
  __shared__ float red[8];
  const int row = blockIdx.x, tid = threadIdx.x;
  const int lane = tid & 31, wv = tid >> 5;
  const size_t base = (size_t)row * DM;
  float v4[4];
  float s = 0.0f;
#pragma unroll
  for (int q = 0; q < 4; ++q) {
    const int c = q * 256 + tid;
    const float v = wres[base + c] + ao[base + c];
    v4[q] = v;
    s += v;
  }
#pragma unroll
  for (int off = 16; off > 0; off >>= 1) s += __shfl_xor(s, off, 32);
  if (lane == 0) red[wv] = s;
  __syncthreads();
  float tot = 0.0f;
#pragma unroll
  for (int q = 0; q < 8; ++q) tot += red[q];
  const float mu = tot * (1.0f / 1024.0f);
  __syncthreads();
  float s2 = 0.0f;
#pragma unroll
  for (int q = 0; q < 4; ++q) { const float d = v4[q] - mu; s2 += d * d; }
#pragma unroll
  for (int off = 16; off > 0; off >>= 1) s2 += __shfl_xor(s2, off, 32);
  if (lane == 0) red[wv] = s2;
  __syncthreads();
  float tot2 = 0.0f;
#pragma unroll
  for (int q = 0; q < 8; ++q) tot2 += red[q];
  const float rstd = rsqrtf(tot2 * (1.0f / 1024.0f) + 1e-5f);
#pragma unroll
  for (int q = 0; q < 4; ++q) {
    const int c = q * 256 + tid;
    out[base + c] = (v4[q] - mu) * rstd * gamma[c] + beta[c];
  }
}

// ---------------------------------------------------------------------------
// Host launcher
// ---------------------------------------------------------------------------
extern "C" void kernel_launch(void* const* d_in, const int* in_sizes, int n_in,
                              void* d_out, int out_size, void* d_ws, size_t ws_size,
                              hipStream_t stream) {
  (void)in_sizes; (void)n_in; (void)out_size; (void)ws_size;
  const float* w     = (const float*)d_in[0];
  const float* r     = (const float*)d_in[1];
  const float* mems  = (const float*)d_in[2];
  // d_in[3] = attn_mask (causal mask computed analytically in-kernel)
  const float* qkvk  = (const float*)d_in[4];
  const float* rk    = (const float*)d_in[5];
  const float* ok    = (const float*)d_in[6];
  const float* rwb   = (const float*)d_in[7];
  const float* rrb   = (const float*)d_in[8];
  const float* gamma = (const float*)d_in[9];
  const float* beta  = (const float*)d_in[10];

  char* base = (char*)d_ws;
  size_t off = 0;
  auto alloc = [&](size_t bytes) -> void* {
    off = (off + 255) & ~(size_t)255;
    void* p = base + off;
    off += bytes;
    return p;
  };

  unsigned short* catB   = (unsigned short*)alloc(8192ull * 1024 * 2);
  unsigned short* qkvWt  = (unsigned short*)alloc(3072ull * 1024 * 2);  // (n,k)
  unsigned short* rB     = (unsigned short*)alloc(2048ull * 1024 * 2);
  unsigned short* rkWt   = (unsigned short*)alloc(1024ull * 1024 * 2);  // (n,k)
  unsigned short* oWt    = (unsigned short*)alloc(1024ull * 1024 * 2);  // (n,k)
  unsigned short* wheads = (unsigned short*)alloc(8192ull * 3072 * 2);
  unsigned short* rhk    = (unsigned short*)alloc(2048ull * 1024 * 2);
  unsigned short* Qrw    = (unsigned short*)alloc(64ull * 1024 * 64 * 2);
  unsigned short* Qrr    = (unsigned short*)alloc(64ull * 1024 * 64 * 2);
  unsigned short* Kn     = (unsigned short*)alloc(64ull * 2048 * 64 * 2);
  unsigned short* Vt     = (unsigned short*)alloc(64ull * 64 * 2048 * 2);
  unsigned short* avec   = (unsigned short*)alloc(4096ull * 1024 * 2);
  float*          aout   = (float*)alloc(4096ull * 1024 * 4);

  // cat = [mems; w] flattens contiguously in (len, bsz, dm)
  k_f32_to_bf16<<<4096, 256, 0, stream>>>(mems, catB, 1048576);
  k_f32_to_bf16<<<4096, 256, 0, stream>>>(w, catB + 4194304, 1048576);
  k_f32_to_bf16<<<2048, 256, 0, stream>>>(r, rB, 524288);
  // weights -> bf16, transposed to N-major (WMMA-B-ready)
  k_f32_to_bf16_t<<<12288, 256, 0, stream>>>(qkvk, qkvWt, 1024, 3072);
  k_f32_to_bf16_t<<<4096, 256, 0, stream>>>(rk, rkWt, 1024, 1024);
  k_f32_to_bf16_t<<<4096, 256, 0, stream>>>(ok, oWt, 1024, 1024);

  k_gemm_bf16<3072, 1024, false><<<dim3(24, 64), 256, 0, stream>>>(catB, qkvWt, wheads);
  k_gemm_bf16<1024, 1024, false><<<dim3(8, 16), 256, 0, stream>>>(rB, rkWt, rhk);

  k_repack<<<32768, 256, 0, stream>>>(wheads, rwb, rrb, Qrw, Qrr, Kn, Vt);

  k_flash_attn<<<dim3(16, 16, 4), 128, 0, stream>>>(Qrw, Qrr, Kn, Vt, rhk, avec);

  k_gemm_bf16<1024, 1024, true><<<dim3(8, 32), 256, 0, stream>>>(avec, oWt, aout);

  k_ln<<<4096, 256, 0, stream>>>(w, aout, gamma, beta, (float*)d_out);
}
